// adapt_gcn_operation_29119878267032
// MI455X (gfx1250) — matrix operand order repaired and checked by reading, not compiled, not run on hardware
//
#include <hip/hip_runtime.h>

// ---- types -----------------------------------------------------------------
typedef __attribute__((ext_vector_type(16))) __bf16        v16bf;
typedef __attribute__((ext_vector_type(2)))  __bf16        v2bf;
typedef __attribute__((ext_vector_type(8)))  float         v8f;
typedef __attribute__((ext_vector_type(4)))  float         v4f;
typedef __attribute__((ext_vector_type(4)))  unsigned int  v4u;
typedef __attribute__((ext_vector_type(2)))  unsigned int  v2u;

union Frag {
  v16bf          v;      // WMMA operand (16 bf16 per lane = 8 VGPRs)
  unsigned short s[16];
  v4u            q[2];   // two 16-byte chunks for b128 loads
};

// round-to-nearest-even f32 -> bf16 (bit-level fallback)
__device__ __forceinline__ unsigned short f2bf(float f) {
  union { float f; unsigned u; } x; x.f = f;
  unsigned r = x.u + 0x7FFFu + ((x.u >> 16) & 1u);
  return (unsigned short)(r >> 16);
}

// pack two f32 -> packed bf16 pair (lo = a, hi = b)
__device__ __forceinline__ unsigned pk_bf16(float a, float b) {
#if __has_builtin(__builtin_amdgcn_cvt_pk_bf16_f32)
  union { v2bf v; unsigned u; } c;
  c.v = __builtin_amdgcn_cvt_pk_bf16_f32(a, b);
  return c.u;
#else
  return (unsigned)f2bf(a) | ((unsigned)f2bf(b) << 16);
#endif
}

#define N_NODES 4096
#define C_IN    64
#define C_OUT   64
#define BATCH   8
#define KT      64            // K-tile per barrier (2 WMMA k-steps)

// ---- pre-pass: x [8,4096,64] f32 -> xbfT [8,64,4096] bf16 (transposed) -----
__global__ __launch_bounds__(256) void cvt_x(const float* __restrict__ x,
                                             unsigned short* __restrict__ xbfT) {
  int idx = blockIdx.x * 256 + threadIdx.x;       // 0 .. 8*4096*64-1
  int c = idx & 63;
  int m = (idx >> 6) & (N_NODES - 1);
  int b = idx >> 18;
  xbfT[((b * 64 + c) << 12) + m] = f2bf(x[idx]);  // coalesced read, strided write
}

// ---- pre-pass: W [64,64] f32 -> wbf bf16 (same layout) ---------------------
__global__ __launch_bounds__(256) void cvt_w(const float* __restrict__ W,
                                             unsigned short* __restrict__ wbf) {
  int idx = blockIdx.x * 256 + threadIdx.x;
  wbf[idx] = f2bf(W[idx]);
}

// One block: 16 rows of d_a, all 8 batches x 64 channels. wave w == batch w.
__global__ __launch_bounds__(256) void gcn_fused(
    const unsigned short* __restrict__ xbfT,  // [8,64,4096] bf16
    const float* __restrict__ adj,            // [4096,4096]
    const float* __restrict__ mask,           // [4096,4096]
    const unsigned short* __restrict__ wbf,   // [64,64] bf16
    const float* __restrict__ bias,           // [64]
    float* __restrict__ out) {                // [8,4096,64]

  __shared__ __align__(16) unsigned short aLds[2][16 * KT];     // dbl-buffered
  __shared__ float partials[256];
  __shared__ float invLds[16];
  __shared__ __align__(16) unsigned short oLds[BATCH * 16 * 64]; // bf16 out1

  const int tid  = threadIdx.x;
  const int lane = tid & 31;
  const int wave = tid >> 5;          // batch index
  const int col  = lane & 15;         // tile column / A-frag row
  const int hi   = lane >> 4;         // lane-half select
  const int kbA  = hi * 16;           // A-operand contiguous k base
  const int kbB  = hi * 16;           // B-operand contiguous k base

  const int row0 = blockIdx.x * 16;

  // cooperative a-tile fill: thread -> (row tr, k quad tk4)
  const int tr  = tid >> 4;           // 0..15
  const int tk4 = (tid & 15) * 4;     // 0,4,..,60

  const size_t arow = (size_t)(row0 + tr) * N_NODES + tk4;

  // per-wave x pointers: column (b, nt*16+col), contiguous in k
  const unsigned short* xp[4];
#pragma unroll
  for (int nt = 0; nt < 4; ++nt)
    xp[nt] = xbfT + (((wave * 64 + nt * 16 + col) << 12) + kbB);

  v8f zero = {};
  v8f acc[4];
  acc[0] = zero; acc[1] = zero; acc[2] = zero; acc[3] = zero;

  float partial = 0.f;                // f32 rowsum share (for deg)

  // prologue: fetch first adj/mask quad (streaming -> non-temporal)
  v4f av = __builtin_nontemporal_load(reinterpret_cast<const v4f*>(adj  + arow));
  v4f mv = __builtin_nontemporal_load(reinterpret_cast<const v4f*>(mask + arow));

#pragma unroll 2
  for (int kt = 0; kt < N_NODES; kt += KT) {
    const int buf = (kt >> 6) & 1;
    {
      float p0 = av.x * mv.x, p1 = av.y * mv.y;
      float p2 = av.z * mv.z, p3 = av.w * mv.w;
      partial += (p0 + p1) + (p2 + p3);
      v2u pk;
      pk.x = pk_bf16(p0, p1);
      pk.y = pk_bf16(p2, p3);
      *reinterpret_cast<v2u*>(&aLds[buf][tr * KT + tk4]) = pk;
    }
    if (kt + KT < N_NODES) {          // issue next tile early: overlaps WMMAs
      av = __builtin_nontemporal_load(
          reinterpret_cast<const v4f*>(adj  + arow + kt + KT));
      mv = __builtin_nontemporal_load(
          reinterpret_cast<const v4f*>(mask + arow + kt + KT));
    }
    __syncthreads();                  // single barrier: dbl-buffer makes it safe

#pragma unroll
    for (int ks = 0; ks < 2; ++ks) {  // two 32-wide k-steps
      const int kk = ks * 32;
      Frag fa;                        // A: row=col, k = kk+kbA{+0..15}
      const unsigned short* ap = &aLds[buf][col * KT + kk + kbA];
      fa.q[0] = *reinterpret_cast<const v4u*>(ap);
      fa.q[1] = *reinterpret_cast<const v4u*>(ap + 8);
#pragma unroll
      for (int nt = 0; nt < 4; ++nt) {
        Frag fb;                      // B: 32 contiguous bytes of xbfT (L2-hot)
        const unsigned short* bp = xp[nt] + kt + kk;
        fb.q[0] = *reinterpret_cast<const v4u*>(bp);
        fb.q[1] = *reinterpret_cast<const v4u*>(bp + 8);
        acc[nt] = __builtin_amdgcn_wmma_f32_16x16x32_bf16(
            false, fa.v, false, fb.v, (short)0, acc[nt], false, false);
      }
    }
  }

  // ---- deg -> inv (f32; 1/0 = inf -> 0 per reference) ----------------------
  partials[tid] = partial;
  __syncthreads();
  if (tid < 16) {
    float s = 0.f;
#pragma unroll
    for (int j = 0; j < 16; ++j) s += partials[tid * 16 + j];
    invLds[tid] = (s != 0.f) ? (1.f / s) : 0.f;
  }
  __syncthreads();

  float inv8[8];
#pragma unroll
  for (int i = 0; i < 8; ++i) inv8[i] = invLds[hi * 8 + i];

  // scale by inv[row], round to bf16, park in LDS for 2nd GEMM
  unsigned short* op = &oLds[wave * (16 * 64)];
#pragma unroll
  for (int nt = 0; nt < 4; ++nt)
#pragma unroll
    for (int i = 0; i < 8; i += 2) {  // C-layout: vgpr i, half hi -> row i+8*hi
      op[(i + 0 + 8 * hi) * 64 + nt * 16 + col] = f2bf(acc[nt][i + 0] * inv8[i + 0]);
      op[(i + 1 + 8 * hi) * 64 + nt * 16 + col] = f2bf(acc[nt][i + 1] * inv8[i + 1]);
    }
  __syncthreads();

  // ---- epilogue GEMM: out1[16x64] @ W^T (K=64 -> 2 WMMA steps) -------------
  Frag fa2[2];
  const unsigned short* o2 = &oLds[wave * (16 * 64) + col * 64];
#pragma unroll
  for (int ks = 0; ks < 2; ++ks) {
    fa2[ks].q[0] = *reinterpret_cast<const v4u*>(o2 + ks * 32 + kbA);
    fa2[ks].q[1] = *reinterpret_cast<const v4u*>(o2 + ks * 32 + kbA + 8);
  }

  float* outp = out + (size_t)wave * (N_NODES * C_OUT);
#pragma unroll
  for (int nt = 0; nt < 4; ++nt) {
    const int o = nt * 16 + col;
    v8f acc2 = zero;
#pragma unroll
    for (int ks = 0; ks < 2; ++ks) {
      Frag fw;                        // B: W^T[c,o] = wbf[o][c], c contiguous
      const unsigned short* wp = wbf + o * C_IN + ks * 32 + kbB;
      fw.q[0] = *reinterpret_cast<const v4u*>(wp);
      fw.q[1] = *reinterpret_cast<const v4u*>(wp + 8);
      acc2 = __builtin_amdgcn_wmma_f32_16x16x32_bf16(
          false, fa2[ks].v, false, fw.v, (short)0, acc2, false, false);
    }
    const float bo = bias[o];
#pragma unroll
    for (int i = 0; i < 8; ++i) {
      float r = acc2[i] + bo;
      r = r > 0.f ? r : 0.f;          // relu
      __builtin_nontemporal_store(
          r, &outp[(size_t)(row0 + i + 8 * hi) * C_OUT + o]);
    }
  }
}

extern "C" void kernel_launch(void* const* d_in, const int* in_sizes, int n_in,
                              void* d_out, int out_size, void* d_ws, size_t ws_size,
                              hipStream_t stream) {
  const float* x    = (const float*)d_in[0];
  const float* adj  = (const float*)d_in[1];
  const float* mask = (const float*)d_in[2];
  const float* W    = (const float*)d_in[3];
  const float* b    = (const float*)d_in[4];
  float* out = (float*)d_out;

  unsigned short* xbfT = (unsigned short*)d_ws;                       // 4 MB
  unsigned short* wbf  = (unsigned short*)((char*)d_ws + (size_t)BATCH * C_IN * N_NODES * 2);

  cvt_x<<<(BATCH * N_NODES * C_IN) / 256, 256, 0, stream>>>(x, xbfT);
  cvt_w<<<(C_OUT * C_IN) / 256, 256, 0, stream>>>(W, wbf);
  gcn_fused<<<N_NODES / 16, 256, 0, stream>>>(xbfT, adj, mask, wbf, b, out);
}
